// TinyMoE_16767552324179
// MI455X (gfx1250) — matrix operand order repaired and checked by reading, not compiled, not run on hardware
//
#include <hip/hip_runtime.h>
#include <hip/hip_bf16.h>

typedef __attribute__((ext_vector_type(16))) __bf16 v16bf;
typedef __attribute__((ext_vector_type(2)))  __bf16 v2bf;
typedef __attribute__((ext_vector_type(8)))  float  v8f;

union Frag { uint4 u4[2]; unsigned int u[8]; v16bf b; };
union Acc  { v8f v; float f[8]; };

// ---- bf16 conversion: hardware v_cvt_pk_bf16_f32 (RNE), 1 VALU op ----------
#if __has_builtin(__builtin_amdgcn_cvt_pk_bf16_f32)
__device__ __forceinline__ unsigned int pack_bf16(float lo, float hi) {
    v2bf p = __builtin_amdgcn_cvt_pk_bf16_f32(lo, hi);
    return __builtin_bit_cast(unsigned int, p);
}
#else
__device__ __forceinline__ unsigned int pack_bf16(float lo, float hi) {
    unsigned int r;
    asm("v_cvt_pk_bf16_f32 %0, %1, %2" : "=v"(r) : "v"(lo), "v"(hi));
    return r;
}
#endif
__device__ __forceinline__ unsigned short bf16_1(float f) {
    return (unsigned short)(pack_bf16(f, 0.0f) & 0xFFFFu);
}

__device__ __forceinline__ v8f wmma_bf16(const Frag& a, const Frag& b, v8f c) {
    return __builtin_amdgcn_wmma_f32_16x16x32_bf16(false, a.b, false, b.b,
                                                   (short)0, c, false, false);
}

// ---------------------------------------------------------------------------
// Workspace layout (dwords):
//   w1p: [e][kk16][n4][lane32][v8]   = 65536 dwords  (bf16 B-fragments, K=512,N=64)
//   w2p: [e][n32][kk2][lane32][v8]   = 65536 dwords  (bf16 B-fragments, K=64,N=512)
//   rwp: [kk16][lane32][v8]          =  4096 dwords  (router_w padded to N=16)
// B-fragment layout (16-bit B, 32x16): lanes 0-15: col n=lane, v0..3 hold K=2v,2v+1, v4..7 K=8+2v,8+2v+1;
// lanes 16-31: col n=lane-16, K=8+2v,8+2v+1 (v0..3), 16+2v,16+2v+1 (v4..7).
// ---------------------------------------------------------------------------
#define W1P_DW 65536
#define W2P_DW 65536
#define RWP_DW 4096
#define TOTAL_PACK (W1P_DW + W2P_DW + RWP_DW)

__global__ void moe_pack_kernel(const float* __restrict__ rw,
                                const float* __restrict__ w1,
                                const float* __restrict__ w2,
                                unsigned int* __restrict__ ws) {
    int t = blockIdx.x * blockDim.x + threadIdx.x;
    if (t < W1P_DW) {
        int v = t & 7, l = (t >> 3) & 31, n = (t >> 8) & 3, kk = (t >> 10) & 15, e = t >> 14;
        int k0 = kk * 32 + ((v >> 2) << 4) + ((l >> 4) << 3) + 2 * (v & 3);
        int nn = n * 16 + (l & 15);
        float lo = w1[(size_t)(e * 512 + k0) * 64 + nn];
        float hi = w1[(size_t)(e * 512 + k0 + 1) * 64 + nn];
        ws[t] = pack_bf16(lo, hi);
    } else if (t < W1P_DW + W2P_DW) {
        int u = t - W1P_DW;
        int v = u & 7, l = (u >> 3) & 31, kk = (u >> 8) & 1, n = (u >> 9) & 31, e = u >> 14;
        int k0 = kk * 32 + ((v >> 2) << 4) + ((l >> 4) << 3) + 2 * (v & 3);
        int nn = n * 16 + (l & 15);
        float lo = w2[(size_t)(e * 64 + k0) * 512 + nn];
        float hi = w2[(size_t)(e * 64 + k0 + 1) * 512 + nn];
        ws[t] = pack_bf16(lo, hi);
    } else if (t < TOTAL_PACK) {
        int u = t - (W1P_DW + W2P_DW);
        int v = u & 7, l = (u >> 3) & 31, kk = u >> 8;
        int k0 = kk * 32 + ((v >> 2) << 4) + ((l >> 4) << 3) + 2 * (v & 3);
        int nn = l & 15;
        unsigned int r = 0u;                       // pad cols 4..15 with zero
        if (nn < 4) r = pack_bf16(rw[k0 * 4 + nn], rw[(k0 + 1) * 4 + nn]);
        ws[t] = r;
    }
}

// ---------------------------------------------------------------------------
// Main kernel: 4 waves / block, 1 tile of 16 tokens per wave.
// x A-fragments live in VGPRs (128 regs); LDS only for the cross-lane H
// transpose and the router weights.
// ---------------------------------------------------------------------------
struct WaveLds {
    unsigned int hfrag[4][2][32][8]; // gelu(xW1+b1)*w_router, A-frag layout (8 KB)
    float        wrow[16][4];        // softmax router weights per token
};

__launch_bounds__(128)
__global__ void moe_main_kernel(const float* __restrict__ x,
                                const float* __restrict__ rb,
                                const float* __restrict__ b1,
                                const float* __restrict__ b2,
                                const unsigned int* __restrict__ ws,
                                float* __restrict__ out) {
    __shared__ WaveLds lds[4];
    const unsigned int* w1p = ws;
    const unsigned int* w2p = ws + W1P_DW;
    const unsigned int* rwp = ws + W1P_DW + W2P_DW;

    const int tid  = threadIdx.x;
    const int lane = tid & 31;
    const int wv   = tid >> 5;
    WaveLds& L = lds[wv];

    const int tile = blockIdx.x * 4 + wv;        // 2048 tiles total
    const int tok0 = tile * 16;
    const int mr   = lane & 15;                  // token row (A) / column (B,C)
    const int hi   = lane >> 4;

    // ---- Load X tile into registers as bf16 A-fragments ---------------------
    // A-layout 16x32: lane l: row m=l&15; v0..3 hold K=8*hi+{0..7}, v4..7 hold
    // K=16+8*hi+{0..7} (pairs packed per dword).
    Frag xf[16];
    {
        const float4* xr = (const float4*)(x + (size_t)(tok0 + mr) * 512);
        #pragma unroll
        for (int kk = 0; kk < 16; ++kk) {
            int b4 = kk * 8 + 2 * hi;
            float4 a0 = xr[b4],     a1 = xr[b4 + 1];
            float4 c0 = xr[b4 + 4], c1 = xr[b4 + 5];
            xf[kk].u[0] = pack_bf16(a0.x, a0.y);
            xf[kk].u[1] = pack_bf16(a0.z, a0.w);
            xf[kk].u[2] = pack_bf16(a1.x, a1.y);
            xf[kk].u[3] = pack_bf16(a1.z, a1.w);
            xf[kk].u[4] = pack_bf16(c0.x, c0.y);
            xf[kk].u[5] = pack_bf16(c0.z, c0.w);
            xf[kk].u[6] = pack_bf16(c1.x, c1.y);
            xf[kk].u[7] = pack_bf16(c1.z, c1.w);
        }
    }

    // ---- Router logits via WMMA (router_w padded to N=16) -------------------
    {
        Acc r;
        #pragma unroll
        for (int q = 0; q < 8; ++q) r.f[q] = 0.0f;
        #pragma unroll
        for (int kk = 0; kk < 16; ++kk) {
            Frag B; const uint4* bp = (const uint4*)(rwp + (size_t)(kk * 32 + lane) * 8);
            B.u4[0] = bp[0]; B.u4[1] = bp[1];
            r.v = wmma_bf16(xf[kk], B, r.v);
        }
        if ((lane & 15) < 4) {
            float rbv = rb[lane & 15];
            #pragma unroll
            for (int q = 0; q < 8; ++q)
                L.wrow[8 * hi + q][lane & 15] = r.f[q] + rbv;
        }
    }
    __syncthreads();
    if (lane < 16) {                                 // softmax over E=4 per token
        float4 w4 = *(float4*)&L.wrow[lane][0];
        float mx = fmaxf(fmaxf(w4.x, w4.y), fmaxf(w4.z, w4.w));
        float e0 = __expf(w4.x - mx), e1 = __expf(w4.y - mx);
        float e2 = __expf(w4.z - mx), e3 = __expf(w4.w - mx);
        float s = 1.0f / (e0 + e1 + e2 + e3);
        *(float4*)&L.wrow[lane][0] = make_float4(e0 * s, e1 * s, e2 * s, e3 * s);
    }
    __syncthreads();

    // ---- Layer 1 per expert: H = gelu(X@W1 + b1) * w_router -----------------
    for (int e = 0; e < 4; ++e) {
        Acc acc[4];
        #pragma unroll
        for (int n = 0; n < 4; ++n)
            #pragma unroll
            for (int q = 0; q < 8; ++q) acc[n].f[q] = 0.0f;

        const unsigned int* w1e = w1p + (size_t)e * 16384;
        #pragma unroll
        for (int kk = 0; kk < 16; ++kk) {
            #pragma unroll
            for (int n = 0; n < 4; ++n) {
                Frag B; const uint4* bp = (const uint4*)(w1e + (size_t)((kk * 4 + n) * 32 + lane) * 8);
                B.u4[0] = bp[0]; B.u4[1] = bp[1];
                acc[n].v = wmma_bf16(xf[kk], B, acc[n].v);
            }
        }
        // Epilogue: bias + exact GELU + router scale, scatter into A-frag layout.
        float wr[8];
        #pragma unroll
        for (int q = 0; q < 8; ++q) wr[q] = L.wrow[8 * hi + q][e];
        #pragma unroll
        for (int n = 0; n < 4; ++n) {
            float b1v = b1[e * 64 + n * 16 + mr];
            int k    = n * 16 + mr;          // K position of this column in H
            int kkp  = k >> 5;
            int kl   = k & 31;
            int lgrp = ((kl >> 3) & 1) << 4;
            int vp   = ((kl >> 4) << 2) + ((kl & 7) >> 1);
            int half = k & 1;
            #pragma unroll
            for (int q = 0; q < 8; ++q) {
                float v0 = acc[n].f[q] + b1v;
                float g  = 0.5f * v0 * (1.0f + erff(v0 * 0.70710678f)) * wr[q];
                unsigned short* hp =
                    (unsigned short*)&L.hfrag[e][kkp][(8 * hi + q) + lgrp][vp];
                hp[half] = bf16_1(g);
            }
        }
    }
    __syncthreads();

    // ---- Layer 2: out = sum_e (w*H_e) @ W2_e + sum_e w_e*b2_e ---------------
    Frag ha[8];                                      // cache all H A-fragments
    #pragma unroll
    for (int e = 0; e < 4; ++e)
        #pragma unroll
        for (int kk = 0; kk < 2; ++kk) {
            const uint4* hp = (const uint4*)&L.hfrag[e][kk][lane][0];
            ha[e * 2 + kk].u4[0] = hp[0];
            ha[e * 2 + kk].u4[1] = hp[1];
        }
    float4 wr4[8];
    #pragma unroll
    for (int q = 0; q < 8; ++q) wr4[q] = *(float4*)&L.wrow[8 * hi + q][0];

    for (int n = 0; n < 32; ++n) {
        float b20 = b2[n * 16 + mr];
        float b21 = b2[512 + n * 16 + mr];
        float b22 = b2[1024 + n * 16 + mr];
        float b23 = b2[1536 + n * 16 + mr];
        Acc acc;
        #pragma unroll
        for (int q = 0; q < 8; ++q)      // seed accumulator with weighted bias
            acc.f[q] = wr4[q].x * b20 + wr4[q].y * b21 + wr4[q].z * b22 + wr4[q].w * b23;
        #pragma unroll
        for (int e = 0; e < 4; ++e)
            #pragma unroll
            for (int kk = 0; kk < 2; ++kk) {
                Frag B;
                const uint4* bp = (const uint4*)(w2p + (size_t)(((e * 32 + n) * 2 + kk) * 32 + lane) * 8);
                B.u4[0] = bp[0]; B.u4[1] = bp[1];
                acc.v = wmma_bf16(ha[e * 2 + kk], B, acc.v);
            }
        float* orow = out + (size_t)tok0 * 512 + n * 16 + mr;
        #pragma unroll
        for (int q = 0; q < 8; ++q)
            orow[(size_t)(8 * hi + q) * 512] = acc.f[q];
    }
}

// ---------------------------------------------------------------------------
extern "C" void kernel_launch(void* const* d_in, const int* in_sizes, int n_in,
                              void* d_out, int out_size, void* d_ws, size_t ws_size,
                              hipStream_t stream) {
    const float* x  = (const float*)d_in[0];
    const float* rw = (const float*)d_in[1];
    const float* rb = (const float*)d_in[2];
    const float* w1 = (const float*)d_in[3];
    const float* b1 = (const float*)d_in[4];
    const float* w2 = (const float*)d_in[5];
    const float* b2 = (const float*)d_in[6];
    unsigned int* ws = (unsigned int*)d_ws;
    float* out = (float*)d_out;

    // Pack weights to bf16 fragment layout (runs every call; deterministic).
    moe_pack_kernel<<<TOTAL_PACK / 256, 256, 0, stream>>>(rw, w1, w2, ws);

    // 32768 tokens / 16 per wave / 4 waves per block = 512 blocks.
    moe_main_kernel<<<512, 128, 0, stream>>>(x, rb, b1, b2, ws, out);
}
